// PointConvSetAbstraction_38783554683057
// MI455X (gfx1250) — compile-verified, matrix operand order checked by reading, not run on hardware
//
#include <hip/hip_runtime.h>
#include <hip/hip_bf16.h>
#include <math.h>

// ---------------------------------------------------------------------------
// PointConv Set Abstraction for MI455X (gfx1250, wave32, WMMA).
// Pipeline: FPS -> kNN -> [gather+conv1 WMMA] -> BN -> [conv2 WMMA] -> BN
//           -> WeightNet (VALU, tiny channels) -> [aggregate WMMA]
//           -> [final 2048->128 WMMA GEMM] -> BN -> relu -> transpose out.
// GEMM waves own one M-tile and iterate all N-tiles (A-fragment reuse).
// B fragments are staged block-wide into LDS via GLOBAL_LOAD_ASYNC_TO_LDS
// (ASYNCcnt / s_wait_asynccnt) when the toolchain exposes the builtins,
// falling back to load+ds_store otherwise; WMMA B operands then come from
// ds_load_b128. GEMM3 double-buffers the 8KB/K-step B stream.
// Workspace requirement: ~164 MB (offsets below).
// ---------------------------------------------------------------------------

typedef __attribute__((ext_vector_type(16))) _Float16 v16h;
typedef __attribute__((ext_vector_type(8)))  _Float16 v8h;
typedef __attribute__((ext_vector_type(8)))  float    v8f;
typedef __attribute__((ext_vector_type(4)))  int      v4i;
typedef _Float16 f16;

#define B_   8
#define N_   4096
#define D_   64
#define S_   1024
#define K_   32
#define Q_   (B_*S_*K_)     /* 262144 */
#define P_   (B_*S_)        /* 8192   */
#define EPS_ 1e-5f

#if __has_builtin(__builtin_amdgcn_global_load_async_to_lds_b128) && \
    __has_builtin(__builtin_amdgcn_s_wait_asynccnt)
#define HAS_ASYNC 1
#else
#define HAS_ASYNC 0
#endif

// ------------------------- workspace layout (bytes) ------------------------
#define OFF_NIDX   ((size_t)0)                         /* Q ints             */
#define OFF_OFF    (OFF_NIDX   + (size_t)Q_*4)         /* Q*3  f32           */
#define OFF_Z1     (OFF_OFF    + (size_t)Q_*3*4)       /* Q*64 f16 row-major */
#define OFF_Z2T    (OFF_Z1     + (size_t)Q_*64*2)      /* P*128*32 f16 [p][c][k] */
#define OFF_WZ0    (OFF_Z2T    + (size_t)Q_*128*2)     /* Q*8  f16           */
#define OFF_WZ1    (OFF_WZ0    + (size_t)Q_*8*2)       /* Q*8  f16           */
#define OFF_WZ2T   (OFF_WZ1    + (size_t)Q_*8*2)       /* P*16*32 f16 [p][w][k] */
#define OFF_FEATS  (OFF_WZ2T   + (size_t)Q_*16*2)      /* P*2048 f16         */
#define OFF_Y      (OFF_FEATS  + (size_t)P_*2048*2)    /* P*128 f32          */
#define OFF_PT     (OFF_Y      + (size_t)P_*128*4)     /* B*N*64 f16 [b][n][d] */
#define OFF_WT1    (OFF_PT     + (size_t)B_*N_*64*2)   /* 8  frags x512 f16  */
#define OFF_WT2    (OFF_WT1    + (size_t)64*64*2)      /* 16 frags x512 f16  */
#define OFF_LINT   (OFF_WT2    + (size_t)64*128*2)     /* 512 frags x512 f16 */
#define OFF_STATS  (OFF_LINT   + (size_t)2048*128*2)   /* 352 sums + 352 sq  */
#define OFF_PARAMS (OFF_STATS  + (size_t)704*4)        /* 352 sc + 352 sh    */

// per-layer channel offsets inside stats/params
#define CH_L1  0
#define CH_L2  64
#define CH_W0  192
#define CH_W1  200
#define CH_W2  208
#define CH_LIN 224
#define CH_TOT 352

// ---------------------------------------------------------------------------
__device__ __forceinline__ int kmap(int lane, int h) {
    return ((h >> 3) << 4) + ((lane >> 4) << 3) + (h & 7);
}
__device__ __forceinline__ v8f wmma16x16x32(v16h a, v16h b, v8f c) {
    return __builtin_amdgcn_wmma_f32_16x16x32_f16(false, a, false, b,
                                                  (short)0, c, false, false);
}
__device__ __forceinline__ v16h cat8(v8h lo, v8h hi) {
    return __builtin_shufflevector(lo, hi, 0, 1, 2, 3, 4, 5, 6, 7,
                                   8, 9, 10, 11, 12, 13, 14, 15);
}
// A fragment from a K-contiguous f16 row (two b128 loads)
__device__ __forceinline__ v16h load_a_frag(const f16* rowp, int kb, int lane) {
    int k0 = kb * 32 + ((lane >> 4) << 3);
    v8h lo = *(const v8h*)(rowp + k0);
    v8h hi = *(const v8h*)(rowp + k0 + 16);
    return cat8(lo, hi);
}
// B fragment out of an LDS staging buffer (two ds_load_b128)
__device__ __forceinline__ v16h load_b_frag_lds(const f16* lb, int fi, int lane) {
    return *(const v16h*)(lb + fi * 512 + lane * 16);
}
// copy one 16-byte chunk global -> LDS (async on CDNA5, fallback ld+ds_store)
__device__ __forceinline__ void cp16(f16* ldst, const f16* gsrc) {
#if HAS_ASYNC
    __builtin_amdgcn_global_load_async_to_lds_b128(
        (__attribute__((address_space(1))) v4i*)gsrc,
        (__attribute__((address_space(3))) v4i*)ldst, 0, 0);
#else
    *(v8h*)ldst = *(const v8h*)gsrc;
#endif
}
__device__ __forceinline__ void wait_async0() {
#if HAS_ASYNC
    __builtin_amdgcn_s_wait_asynccnt(0);
#endif
}
__device__ __forceinline__ void wait_async2() {
#if HAS_ASYNC
    __builtin_amdgcn_s_wait_asynccnt(2);
#endif
}

// ---------------------------- misc small kernels ---------------------------
__global__ void zero_kernel(float* p, int n) {
    int t = blockIdx.x * 256 + threadIdx.x;
    if (t < n) p[t] = 0.f;
}

// transpose points [B,64,N] f32 -> pt [B,N,64] f16 (coalesced reads)
__global__ void ptrans_kernel(const float* __restrict__ points, f16* __restrict__ pt) {
    int t = blockIdx.x * 256 + threadIdx.x;          // t = (b*64+d)*4096 + n
    if (t >= B_ * 64 * N_) return;
    int b = t >> 18, d = (t >> 12) & 63, n = t & 4095;
    pt[(((size_t)b << 12) + n) * 64 + d] = (f16)points[t];
}

// weights -> f16 B-fragments, frag order [kb][ntile]:
//   frag[fi*512 + lane*16 + h] = W[n][k], fi = kb*NT + nt,
//   k = kb*32 + kmap(lane,h), n = nt*16 + lane%16
__global__ void convw_kernel(const float* __restrict__ c1w,
                             const float* __restrict__ c2w,
                             const float* __restrict__ linw,
                             f16* __restrict__ wT1, f16* __restrict__ wT2,
                             f16* __restrict__ linT) {
    int t = blockIdx.x * 256 + threadIdx.x;
    int fi = t >> 9, r = t & 511, lane = r >> 4, h = r & 15;
    int km = kmap(lane, h), nc = lane & 15;
    if (t < 8 * 512) {     // c1: KB=2, NT=4
        int kb = fi >> 2, nt = fi & 3;
        wT1[t] = (f16)c1w[(nt * 16 + nc) * 64 + kb * 32 + km];
    }
    if (t < 16 * 512) {    // c2: KB=2, NT=8
        int kb = fi >> 3, nt = fi & 7;
        wT2[t] = (f16)c2w[(nt * 16 + nc) * 64 + kb * 32 + km];
    }
    if (t < 512 * 512) {   // lin: KB=64, NT=8
        int kb = fi >> 3, nt = fi & 7;
        linT[t] = (f16)linw[(size_t)(nt * 16 + nc) * 2048 + kb * 32 + km];
    }
}

// BN finalize: scale = g*rsqrt(var+eps), shift = beta - mean*scale
__global__ void bnfin_kernel(const float* __restrict__ sum, const float* __restrict__ ssq,
                             const float* __restrict__ gamma, const float* __restrict__ beta,
                             float* __restrict__ sc, float* __restrict__ sh,
                             int nch, float invcnt) {
    int c = blockIdx.x * blockDim.x + threadIdx.x;
    if (c >= nch) return;
    float m = sum[c] * invcnt;
    float v = ssq[c] * invcnt - m * m;
    float inv = rsqrtf(v + EPS_);
    float s = gamma[c] * inv;
    sc[c] = s;
    sh[c] = beta[c] - m * s;
}

// ------------------------------- FPS ---------------------------------------
__global__ void fps_kernel(const float* __restrict__ xyz,
                           float* __restrict__ out_newxyz /* d_out [B,3,S] */) {
    __shared__ float dist[N_];
    __shared__ float rv[256];
    __shared__ int   ri[256];
    __shared__ int   s_far;
    const int b = blockIdx.x;
    const int t = threadIdx.x;
    const float* xb = xyz + (size_t)b * 3 * N_;
    for (int i = t; i < N_; i += 256) dist[i] = 1e10f;
    if (t == 0) s_far = 0;
    __syncthreads();
    for (int it = 0; it < S_; ++it) {
        int far = s_far;
        if (t < 3) out_newxyz[((size_t)b * 3 + t) * S_ + it] = xb[t * N_ + far];
        float cx = xb[far], cy = xb[N_ + far], cz = xb[2 * N_ + far];
        float bv = -1.0f; int bi = 0;
        for (int i = t; i < N_; i += 256) {
            float dx = xb[i] - cx, dy = xb[N_ + i] - cy, dz = xb[2 * N_ + i] - cz;
            float d  = dx * dx + dy * dy + dz * dz;
            float nd = fminf(dist[i], d);
            dist[i] = nd;
            if (nd > bv || (nd == bv && i < bi)) { bv = nd; bi = i; }
        }
        rv[t] = bv; ri[t] = bi;
        __syncthreads();
        for (int off = 128; off > 0; off >>= 1) {
            if (t < off) {
                float ov = rv[t + off]; int oi = ri[t + off];
                if (ov > rv[t] || (ov == rv[t] && oi < ri[t])) { rv[t] = ov; ri[t] = oi; }
            }
            __syncthreads();
        }
        if (t == 0) s_far = ri[0];
        __syncthreads();
    }
}

// ------------------------------- kNN ---------------------------------------
__global__ void knn_kernel(const float* __restrict__ xyz,
                           const float* __restrict__ newxyz,
                           int* __restrict__ nidx, float* __restrict__ off) {
    __shared__ float sd[2 * N_];
    const int wave = threadIdx.x >> 5;
    const int lane = threadIdx.x & 31;
    const int g = blockIdx.x * 2 + wave;
    const int b = g >> 10, s = g & 1023;
    const float* xb = xyz + (size_t)b * 3 * N_;
    float qx = newxyz[((size_t)b * 3 + 0) * S_ + s];
    float qy = newxyz[((size_t)b * 3 + 1) * S_ + s];
    float qz = newxyz[((size_t)b * 3 + 2) * S_ + s];
    float* d = sd + wave * N_;
    for (int i = lane; i < N_; i += 32) {
        float dx = xb[i] - qx, dy = xb[N_ + i] - qy, dz = xb[2 * N_ + i] - qz;
        d[i] = dx * dx + dy * dy + dz * dz;
    }
    __syncthreads();
    for (int r = 0; r < K_; ++r) {
        float bv = 3.0e38f; int bi = N_;
        for (int i = lane; i < N_; i += 32) {
            float v = d[i];
            if (v < bv) { bv = v; bi = i; }
        }
        for (int m = 16; m > 0; m >>= 1) {
            float ov = __shfl_xor(bv, m, 32);
            int   oi = __shfl_xor(bi, m, 32);
            if (ov < bv || (ov == bv && oi < bi)) { bv = ov; bi = oi; }
        }
        if (lane == 0) {
            nidx[(size_t)g * K_ + r] = bi;
            off[((size_t)g * K_ + r) * 3 + 0] = xb[bi] - qx;
            off[((size_t)g * K_ + r) * 3 + 1] = xb[N_ + bi] - qy;
            off[((size_t)g * K_ + r) * 3 + 2] = xb[2 * N_ + bi] - qz;
            d[bi] = 3.0e38f;
        }
        __syncthreads();
    }
}

// ------------------- GEMM1: gather + conv1 (64 -> 64) ----------------------
// B weights (8KB) staged once into LDS (async); A gathered rows loaded once,
// reused over all 4 N-tiles.
__global__ void gemm1_kernel(const f16* __restrict__ pt,
                             const int* __restrict__ nidx,
                             const f16* __restrict__ wT1, const float* __restrict__ bias,
                             f16* __restrict__ z1,
                             float* __restrict__ sum, float* __restrict__ ssq) {
    __shared__ __align__(32) f16 lb[8 * 512];
    const int tid  = threadIdx.x;
    const int lane = tid & 31;
    for (int i = tid; i < 8 * 64; i += 256) cp16(&lb[i * 8], wT1 + i * 8);
    wait_async0();
    __syncthreads();

    const int mtile = blockIdx.x * 8 + (tid >> 5);           // 16384 tiles
    const int row   = mtile * 16 + (lane & 15);              // q in [0, Q_)
    const int b     = row >> 15;                             // / (S_*K_)
    const int n     = nidx[row];
    const f16* ptb  = pt + ((((size_t)b << 12) + n) << 6);   // [b][n][0..63]
    v16h a0 = load_a_frag(ptb, 0, lane);
    v16h a1 = load_a_frag(ptb, 1, lane);
    v8f acc[4] = {};
#pragma unroll
    for (int nt = 0; nt < 4; ++nt) {
        acc[nt] = wmma16x16x32(a0, load_b_frag_lds(lb, nt,     lane), acc[nt]);
        acc[nt] = wmma16x16x32(a1, load_b_frag_lds(lb, 4 + nt, lane), acc[nt]);
    }
#pragma unroll
    for (int nt = 0; nt < 4; ++nt) {
        const int ncol = nt * 16 + (lane & 15);
        float bv = bias[ncol], ls = 0.f, lq = 0.f;
#pragma unroll
        for (int r = 0; r < 8; ++r) {
            int m = mtile * 16 + r + ((lane >> 4) << 3);
            float z = acc[nt][r] + bv;
            ls += z; lq += z * z;
            z1[(size_t)m * 64 + ncol] = (f16)z;
        }
        atomicAdd(&sum[ncol], ls);
        atomicAdd(&ssq[ncol], lq);
    }
}

// ---------------- GEMM2: BN+relu(z1) @ conv2 (64 -> 128) -------------------
// B weights (16KB) staged once into LDS (async). A loaded + BN-affine'd once,
// reused over 8 N-tiles. Writes z2 transposed per centroid: z2t[p][c][k].
__global__ void gemm2_kernel(const f16* __restrict__ z1,
                             const float* __restrict__ p1s, const float* __restrict__ p1h,
                             const f16* __restrict__ wT2, const float* __restrict__ bias,
                             f16* __restrict__ z2t,
                             float* __restrict__ sum, float* __restrict__ ssq) {
    __shared__ __align__(32) f16 lb[16 * 512];
    const int tid  = threadIdx.x;
    const int lane = tid & 31;
    for (int i = tid; i < 16 * 64; i += 256) cp16(&lb[i * 8], wT2 + i * 8);
    wait_async0();
    __syncthreads();

    const int mtile = blockIdx.x * 8 + (tid >> 5);           // 16384 tiles
    const int row   = mtile * 16 + (lane & 15);
    const f16* z1r  = z1 + (size_t)row * 64;
    v16h afr[2];
#pragma unroll
    for (int kb = 0; kb < 2; ++kb) {
        int k0 = kb * 32 + ((lane >> 4) << 3);
        v8h lo = *(const v8h*)(z1r + k0);
        v8h hi = *(const v8h*)(z1r + k0 + 16);
        v16h a;
#pragma unroll
        for (int j = 0; j < 8; ++j) {
            a[j]     = (f16)fmaxf((float)lo[j] * p1s[k0 + j]      + p1h[k0 + j],      0.f);
            a[j + 8] = (f16)fmaxf((float)hi[j] * p1s[k0 + 16 + j] + p1h[k0 + 16 + j], 0.f);
        }
        afr[kb] = a;
    }
    v8f acc[8] = {};
#pragma unroll
    for (int nt = 0; nt < 8; ++nt) {
        acc[nt] = wmma16x16x32(afr[0], load_b_frag_lds(lb, nt,     lane), acc[nt]);
        acc[nt] = wmma16x16x32(afr[1], load_b_frag_lds(lb, 8 + nt, lane), acc[nt]);
    }
    const int qbase = mtile * 16 + ((lane >> 4) << 3);
    const int p     = qbase >> 5;
    const int kbase = qbase & 31;
#pragma unroll
    for (int nt = 0; nt < 8; ++nt) {
        const int ncol = nt * 16 + (lane & 15);
        float bv = bias[ncol], ls = 0.f, lq = 0.f;
        v8h zs;
#pragma unroll
        for (int r = 0; r < 8; ++r) {
            float z = acc[nt][r] + bv;
            ls += z; lq += z * z;
            zs[r] = (f16)z;
        }
        *(v8h*)(z2t + ((size_t)p * 128 + ncol) * 32 + kbase) = zs;
        atomicAdd(&sum[ncol], ls);
        atomicAdd(&ssq[ncol], lq);
    }
}

// --------------------------- WeightNet (VALU) ------------------------------
__global__ void wn0_kernel(const float* __restrict__ off,
                           const float* __restrict__ W, const float* __restrict__ bias,
                           f16* __restrict__ out,
                           float* __restrict__ sum, float* __restrict__ ssq) {
    int t = blockIdx.x * 256 + threadIdx.x;
    if (t >= Q_ * 8) return;
    int q = t >> 3, c = t & 7;
    float z = bias[c];
    for (int d2 = 0; d2 < 3; ++d2) z += off[(size_t)q * 3 + d2] * W[c * 3 + d2];
    out[t] = (f16)z;
    atomicAdd(&sum[c], z);
    atomicAdd(&ssq[c], z * z);
}

// generic layer; transposeOut=1 stores out[(q/32)*Cout + c][k=q%32]
__global__ void wn_kernel(const f16* __restrict__ in,
                          const float* __restrict__ ps, const float* __restrict__ ph,
                          const float* __restrict__ W, const float* __restrict__ bias,
                          int Cin, int Cout, f16* __restrict__ out,
                          float* __restrict__ sum, float* __restrict__ ssq,
                          int total, int transposeOut) {
    int t = blockIdx.x * 256 + threadIdx.x;
    if (t >= total) return;
    int q = t / Cout, c = t % Cout;
    float z = bias[c];
    for (int d2 = 0; d2 < Cin; ++d2) {
        float x = fmaxf((float)in[(size_t)q * Cin + d2] * ps[d2] + ph[d2], 0.f);
        z += x * W[c * Cin + d2];
    }
    size_t o = transposeOut ? ((size_t)((q >> 5) * Cout + c) * 32 + (q & 31)) : (size_t)t;
    out[o] = (f16)z;
    atomicAdd(&sum[c], z);
    atomicAdd(&ssq[c], z * z);
}

// --------- aggregation: feats[p][c][w] = sum_k f[p,k,c] * w[p,k,w] ---------
__global__ void aggregate_kernel(const f16* __restrict__ z2t,
                                 const float* __restrict__ p2s, const float* __restrict__ p2h,
                                 const f16* __restrict__ wz2t,
                                 const float* __restrict__ pws, const float* __restrict__ pwh,
                                 f16* __restrict__ feats) {
    const int lane = threadIdx.x & 31;
    const int w    = threadIdx.x >> 5;       // channel tile 0..7
    const int p    = blockIdx.x;             // centroid
    const int crow = w * 16 + (lane & 15);
    const int ncol = lane & 15;
    const int k0   = (lane >> 4) << 3;
    const f16* ar = z2t  + ((size_t)p * 128 + crow) * 32;
    const f16* br = wz2t + ((size_t)p * 16  + ncol) * 32;
    v8h alo = *(const v8h*)(ar + k0);
    v8h ahi = *(const v8h*)(ar + k0 + 16);
    v8h blo = *(const v8h*)(br + k0);
    v8h bhi = *(const v8h*)(br + k0 + 16);
    const float sa = p2s[crow], ha = p2h[crow];
    const float sb = pws[ncol], hb = pwh[ncol];
    v16h a, bb;
#pragma unroll
    for (int j = 0; j < 8; ++j) {
        a[j]      = (f16)fmaxf((float)alo[j] * sa + ha, 0.f);
        a[j + 8]  = (f16)fmaxf((float)ahi[j] * sa + ha, 0.f);
        bb[j]     = (f16)fmaxf((float)blo[j] * sb + hb, 0.f);
        bb[j + 8] = (f16)fmaxf((float)bhi[j] * sb + hb, 0.f);
    }
    v8f c = {};
    c = wmma16x16x32(a, bb, c);
#pragma unroll
    for (int r = 0; r < 8; ++r) {
        int cr = w * 16 + r + ((lane >> 4) << 3);
        feats[(size_t)p * 2048 + cr * 16 + ncol] = (f16)c[r];
    }
}

// ------------------ final GEMM: feats (P x 2048) @ linT ---------------------
// Double-buffered async staging of the per-K-step 8KB B block into LDS; each
// wave computes all 8 N-tiles (A read once). Per K-step: 2 A b128 loads,
// 2 async stage issues, 16 ds_load_b128, 8 WMMAs.
__global__ void gemm3_kernel(const f16* __restrict__ feats,
                             const f16* __restrict__ linT, const float* __restrict__ bias,
                             float* __restrict__ y,
                             float* __restrict__ sum, float* __restrict__ ssq) {
    __shared__ __align__(32) f16 lbuf[2][8 * 512];   // 2 x 8KB
    const int tid  = threadIdx.x;
    const int lane = tid & 31;
    const int mtile = blockIdx.x * 8 + (tid >> 5);   // 512 tiles
    const int row   = mtile * 16 + (lane & 15);
    const f16* arow = feats + (size_t)row * 2048;

    auto stage = [&](int kb2, int buf2) {
        const f16* src = linT + (size_t)kb2 * 4096;  // 8 frags x 512 halves
        f16* dst = &lbuf[buf2][0];
        for (int i = tid; i < 512; i += 256)         // 512 chunks of 16B
            cp16(dst + i * 8, src + i * 8);
    };

    stage(0, 0);
    v8f acc[8] = {};
    for (int kb = 0; kb < 64; ++kb) {
        const int buf = kb & 1;
        if (kb < 63) { stage(kb + 1, buf ^ 1); wait_async2(); }
        else         { wait_async0(); }
        __syncthreads();
        v16h a = load_a_frag(arow, kb, lane);
#pragma unroll
        for (int nt = 0; nt < 8; ++nt) {
            acc[nt] = wmma16x16x32(a, load_b_frag_lds(&lbuf[buf][0], nt, lane), acc[nt]);
        }
        __syncthreads();
    }
#pragma unroll
    for (int nt = 0; nt < 8; ++nt) {
        const int ncol = nt * 16 + (lane & 15);
        float bv = bias[ncol], ls = 0.f, lq = 0.f;
#pragma unroll
        for (int r = 0; r < 8; ++r) {
            int m = mtile * 16 + r + ((lane >> 4) << 3);
            float z = acc[nt][r] + bv;
            ls += z; lq += z * z;
            y[(size_t)m * 128 + ncol] = z;
        }
        atomicAdd(&sum[ncol], ls);
        atomicAdd(&ssq[ncol], lq);
    }
}

// ---------------- final BN + relu + transpose to [B,128,S] -----------------
__global__ void out_kernel(const float* __restrict__ y,
                           const float* __restrict__ sc, const float* __restrict__ sh,
                           float* __restrict__ outp) {
    int t = blockIdx.x * 256 + threadIdx.x;
    if (t >= P_ * 128) return;
    int p = t >> 7, ch = t & 127;
    int b = p >> 10, s = p & 1023;
    outp[((size_t)b * 128 + ch) * 1024 + s] = fmaxf(y[t] * sc[ch] + sh[ch], 0.f);
}

// ---------------------------------------------------------------------------
extern "C" void kernel_launch(void* const* d_in, const int* in_sizes, int n_in,
                              void* d_out, int out_size, void* d_ws, size_t ws_size,
                              hipStream_t stream) {
    const float* xyz    = (const float*)d_in[0];
    const float* points = (const float*)d_in[1];
    const float* c1_w   = (const float*)d_in[2];
    const float* c1_b   = (const float*)d_in[3];
    const float* g1     = (const float*)d_in[4];
    const float* b1     = (const float*)d_in[5];
    const float* c2_w   = (const float*)d_in[6];
    const float* c2_b   = (const float*)d_in[7];
    const float* g2     = (const float*)d_in[8];
    const float* b2     = (const float*)d_in[9];
    const float* w0_w   = (const float*)d_in[10];
    const float* w0_b   = (const float*)d_in[11];
    const float* w0_g   = (const float*)d_in[12];
    const float* w0_be  = (const float*)d_in[13];
    const float* w1_w   = (const float*)d_in[14];
    const float* w1_b   = (const float*)d_in[15];
    const float* w1_g   = (const float*)d_in[16];
    const float* w1_be  = (const float*)d_in[17];
    const float* w2_w   = (const float*)d_in[18];
    const float* w2_b   = (const float*)d_in[19];
    const float* w2_g   = (const float*)d_in[20];
    const float* w2_be  = (const float*)d_in[21];
    const float* lin_w  = (const float*)d_in[22];
    const float* lin_b  = (const float*)d_in[23];
    const float* lg     = (const float*)d_in[24];
    const float* lb     = (const float*)d_in[25];

    char* ws = (char*)d_ws;
    int*   nidx  = (int*)  (ws + OFF_NIDX);
    float* off   = (float*)(ws + OFF_OFF);
    f16*   z1    = (f16*)  (ws + OFF_Z1);
    f16*   z2t   = (f16*)  (ws + OFF_Z2T);
    f16*   wz0   = (f16*)  (ws + OFF_WZ0);
    f16*   wz1   = (f16*)  (ws + OFF_WZ1);
    f16*   wz2t  = (f16*)  (ws + OFF_WZ2T);
    f16*   feats = (f16*)  (ws + OFF_FEATS);
    float* ylin  = (float*)(ws + OFF_Y);
    f16*   pt    = (f16*)  (ws + OFF_PT);
    f16*   wT1   = (f16*)  (ws + OFF_WT1);
    f16*   wT2   = (f16*)  (ws + OFF_WT2);
    f16*   linT  = (f16*)  (ws + OFF_LINT);
    float* stats = (float*)(ws + OFF_STATS);
    float* sumA  = stats;
    float* ssqA  = stats + CH_TOT;
    float* prm   = (float*)(ws + OFF_PARAMS);
    float* scA   = prm;
    float* shA   = prm + CH_TOT;

    float* outp   = (float*)d_out;
    float* newxyz = outp;                              // [B,3,S]
    float* yout   = outp + (size_t)B_ * 3 * S_;        // [B,128,S]

    // 0) zero BN accumulators
    zero_kernel<<<(704 + 255) / 256, 256, 0, stream>>>(stats, 704);

    // 1) weight fragments + point transpose
    convw_kernel<<<(512 * 512 + 255) / 256, 256, 0, stream>>>(c1_w, c2_w, lin_w,
                                                              wT1, wT2, linT);
    ptrans_kernel<<<(B_ * 64 * N_) / 256, 256, 0, stream>>>(points, pt);

    // 2) FPS (writes new_xyz straight into d_out, transposed)
    fps_kernel<<<B_, 256, 0, stream>>>(xyz, newxyz);

    // 3) kNN + offsets
    knn_kernel<<<P_ / 2, 64, 0, stream>>>(xyz, newxyz, nidx, off);

    // 4) conv1 (gather fused) + BN stats; finalize
    gemm1_kernel<<<(Q_ / 16) / 8, 256, 0, stream>>>(pt, nidx, wT1, c1_b,
                                                    z1, sumA + CH_L1, ssqA + CH_L1);
    bnfin_kernel<<<1, 64, 0, stream>>>(sumA + CH_L1, ssqA + CH_L1, g1, b1,
                                       scA + CH_L1, shA + CH_L1, 64, 1.f / Q_);

    // 5) conv2 + BN stats; finalize
    gemm2_kernel<<<(Q_ / 16) / 8, 256, 0, stream>>>(z1, scA + CH_L1, shA + CH_L1,
                                                    wT2, c2_b, z2t,
                                                    sumA + CH_L2, ssqA + CH_L2);
    bnfin_kernel<<<1, 128, 0, stream>>>(sumA + CH_L2, ssqA + CH_L2, g2, b2,
                                        scA + CH_L2, shA + CH_L2, 128, 1.f / Q_);

    // 6) WeightNet 3->8->8->16 (VALU)
    wn0_kernel<<<(Q_ * 8) / 256, 256, 0, stream>>>(off, w0_w, w0_b, wz0,
                                                   sumA + CH_W0, ssqA + CH_W0);
    bnfin_kernel<<<1, 32, 0, stream>>>(sumA + CH_W0, ssqA + CH_W0, w0_g, w0_be,
                                       scA + CH_W0, shA + CH_W0, 8, 1.f / Q_);
    wn_kernel<<<(Q_ * 8) / 256, 256, 0, stream>>>(wz0, scA + CH_W0, shA + CH_W0,
                                                  w1_w, w1_b, 8, 8, wz1,
                                                  sumA + CH_W1, ssqA + CH_W1, Q_ * 8, 0);
    bnfin_kernel<<<1, 32, 0, stream>>>(sumA + CH_W1, ssqA + CH_W1, w1_g, w1_be,
                                       scA + CH_W1, shA + CH_W1, 8, 1.f / Q_);
    wn_kernel<<<(Q_ * 16) / 256, 256, 0, stream>>>(wz1, scA + CH_W1, shA + CH_W1,
                                                   w2_w, w2_b, 8, 16, wz2t,
                                                   sumA + CH_W2, ssqA + CH_W2, Q_ * 16, 1);
    bnfin_kernel<<<1, 32, 0, stream>>>(sumA + CH_W2, ssqA + CH_W2, w2_g, w2_be,
                                       scA + CH_W2, shA + CH_W2, 16, 1.f / Q_);

    // 7) weighted aggregation over neighbors -> feats [P, 2048] (f16)
    aggregate_kernel<<<P_, 256, 0, stream>>>(z2t, scA + CH_L2, shA + CH_L2,
                                             wz2t, scA + CH_W2, shA + CH_W2, feats);

    // 8) final linear 2048 -> 128 + BN stats; finalize
    gemm3_kernel<<<(P_ / 16) / 8, 256, 0, stream>>>(feats, linT, lin_b, ylin,
                                                    sumA + CH_LIN, ssqA + CH_LIN);
    bnfin_kernel<<<1, 128, 0, stream>>>(sumA + CH_LIN, ssqA + CH_LIN, lg, lb,
                                        scA + CH_LIN, shA + CH_LIN, 128, 1.f / P_);

    // 9) BN + relu + transpose to [B,128,S]
    out_kernel<<<(P_ * 128) / 256, 256, 0, stream>>>(ylin, scA + CH_LIN, shA + CH_LIN,
                                                     yout);
}
